// MessagePassing_79104707658373
// MI455X (gfx1250) — matrix operand order repaired and checked by reading, not compiled, not run on hardware
//
#include <hip/hip_runtime.h>

typedef __attribute__((ext_vector_type(16))) _Float16 v16h;
typedef __attribute__((ext_vector_type(8)))  float    v8f;

#define NWAVES 8
#define TPB (NWAVES * 32)

union FragBits { uint4 u[2]; v16h v; };

__device__ __forceinline__ float fast_tanh(float x) {
  // tanh(x) = 1 - 2/(exp(2x)+1); exp(2x) = exp2(2*log2(e)*x)
  float e = __builtin_amdgcn_exp2f(x * 2.8853900817779268f);
  float r = __builtin_amdgcn_rcpf(e + 1.0f);
  return 1.0f - 2.0f * r;
}

__device__ __forceinline__ v8f wmma16(v16h a, v16h b, v8f c) {
  return __builtin_amdgcn_wmma_f32_16x16x32_f16(false, a, false, b,
                                                (short)0, c, false, false);
}

__global__ __launch_bounds__(TPB) void mp17_kernel(
    const float* __restrict__ inp,  // [B,5]
    const float* __restrict__ Wf,   // [17]
    const float* __restrict__ bf,   // [17]
    const float* __restrict__ Wm,   // [34,17]
    const float* __restrict__ bm,   // [17]
    const float* __restrict__ Wu,   // [68,17]
    const float* __restrict__ bu,   // [17]
    const float* __restrict__ Wr,   // [85]
    const float* __restrict__ br,   // [1]
    float* __restrict__ out,        // [B]
    int ntiles)
{
  __shared__ __align__(16) _Float16 stage[NWAVES][512];  // 16x32 f16 per wave

  const int lane = threadIdx.x & 31;
  const int wv   = threadIdx.x >> 5;
  const int nl   = lane & 15;          // column (C/D) / row (A) index
  const bool hi  = lane >= 16;
  const int kA   = hi ? 8 : 0;         // A-layout K sub-base per lane half
  const int kB   = hi ? 16 : 0;        // unused; B frags also use kA

  _Float16* st = stage[wv];

  // Zero staging once: cols 17..31 then stay zero forever (A padding safe).
  *reinterpret_cast<uint4*>(st + lane * 16) = make_uint4(0u, 0u, 0u, 0u);
  asm volatile("s_wait_dscnt 0x0" ::: "memory");

  // ---- Static weight B-fragments (32x16 f16 tiles, zero-padded 17->32) ----
  // B layout (16-bit, 32x16): lane l holds col n=l&15; halves p=0..15 hold
  // K = (p<8 ? kA+p : 16+kA+(p-8)), kA = (l<16 ? 0 : 8).
  auto mkB = [&](const float* W, int rowoff, int ntile) -> v16h {
    v16h f;
    const int col = ntile * 16 + nl;
#pragma unroll
    for (int p = 0; p < 16; ++p) {
      const int k = (p < 8) ? kA + p : 16 + kA + (p - 8);
      float v = 0.0f;
      if (k < 17 && col < 17) v = W[(rowoff + k) * 17 + col];
      f[p] = (_Float16)v;
    }
    return f;
  };
  auto mkB2 = [&](const float* W, int r0, int r1, int ntile) -> v16h {
    v16h f;
    const int col = ntile * 16 + nl;
#pragma unroll
    for (int p = 0; p < 16; ++p) {
      const int k = (p < 8) ? kA + p : 16 + kA + (p - 8);
      float v = 0.0f;
      if (k < 17 && col < 17) v = W[(r0 + k) * 17 + col] + W[(r1 + k) * 17 + col];
      f[p] = (_Float16)v;
    }
    return f;
  };

  const v16h Bm1_0 = mkB(Wm, 0, 0),  Bm1_1 = mkB(Wm, 0, 1);    // p-part of msg
  const v16h Bm2_0 = mkB(Wm, 17, 0), Bm2_1 = mkB(Wm, 17, 1);   // q-part of msg
  const v16h Bu1_0 = mkB(Wu, 0, 0),  Bu1_1 = mkB(Wu, 0, 1);    // m1 part
  const v16h Bu2_0 = mkB(Wu, 17, 0), Bu2_1 = mkB(Wu, 17, 1);   // m2 part
  const v16h Buc_0 = mkB2(Wu, 34, 51, 0), Buc_1 = mkB2(Wu, 34, 51, 1); // u==f fused

  // ---- Per-lane scalars ----
  // A layout: lane holds row M=nl; halves p: K = (p<8 ? kA+p : 16+kA+(p-8)).
  // Only halves 0..8 can hit K<17 (half 8 -> K=16 on low lanes only);
  // halves 9..15 are always padding -> exact zero.
  float wfv[9], bfv[9];
#pragma unroll
  for (int p = 0; p < 8; ++p) {
    const int k = kA + p;            // 0..7 (low) / 8..15 (high): always valid
    wfv[p] = Wf[k];
    bfv[p] = bf[k];
  }
  wfv[8] = hi ? 0.0f : Wf[16];
  bfv[8] = hi ? 0.0f : bf[16];

  const float bm0 = bm[nl], bm16 = bm[16];
  const float bu0 = bu[nl], bu16 = bu[16];
  float wr0[5], wr16[5];
#pragma unroll
  for (int i = 0; i < 5; ++i) {
    wr0[i]  = Wr[i * 17 + nl];
    wr16[i] = (nl == 0) ? Wr[i * 17 + 16] : 0.0f;
  }
  const float brv = br[0];

  // Feature fragment: A[M=nl][K] = tanh(inp_scalar * Wf[K] + bf[K]); pad -> 0.
  auto mkFeat = [&](float s) -> v16h {
    v16h f;
#pragma unroll
    for (int p = 0; p < 9; ++p) f[p] = (_Float16)fast_tanh(s * wfv[p] + bfv[p]);
#pragma unroll
    for (int p = 9; p < 16; ++p) f[p] = (_Float16)0.0f;
    return f;
  };

  // D-layout (per lane: rows r+(hi?8:0), col nl) -> staged f16 matrix -> A frag
  auto stageMat = [&](v8f c0, v8f c1) {
    const int rb = hi ? 8 : 0;
#pragma unroll
    for (int r = 0; r < 8; ++r)
      st[(r + rb) * 32 + nl] = (_Float16)fast_tanh(c0[r]);
    if (nl == 0) {
#pragma unroll
      for (int r = 0; r < 8; ++r)
        st[(r + rb) * 32 + 16] = (_Float16)fast_tanh(c1[r]);   // column 16
    }
    asm volatile("s_wait_dscnt 0x0" ::: "memory");
  };
  auto loadA = [&]() -> v16h {
    FragBits fb;
    fb.u[0] = *reinterpret_cast<const uint4*>(st + nl * 32 + kA);       // K lo
    fb.u[1] = *reinterpret_cast<const uint4*>(st + nl * 32 + 16 + kA);  // K hi
    asm volatile("" ::: "memory");
    return fb.v;
  };

  auto message = [&](v16h p, v16h q) -> v16h {
    v8f c0, c1;
#pragma unroll
    for (int r = 0; r < 8; ++r) { c0[r] = bm0; c1[r] = bm16; }
    // Alternate independent accumulators; same-A pairs adjacent.
    c0 = wmma16(p, Bm1_0, c0);
    c1 = wmma16(p, Bm1_1, c1);
    c0 = wmma16(q, Bm2_0, c0);
    c1 = wmma16(q, Bm2_1, c1);
    stageMat(c0, c1);
    return loadA();
  };

  auto update_ro = [&](v16h m1, v16h m2, v16h x, bool hasMsg, int i, float* acc) {
    v8f c0, c1;
#pragma unroll
    for (int r = 0; r < 8; ++r) { c0[r] = bu0; c1[r] = bu16; }
    if (hasMsg) {
      c0 = wmma16(m1, Bu1_0, c0);
      c1 = wmma16(m1, Bu1_1, c1);
      c0 = wmma16(m2, Bu2_0, c0);
      c1 = wmma16(m2, Bu2_1, c1);
    }
    c0 = wmma16(x, Buc_0, c0);
    c1 = wmma16(x, Buc_1, c1);
#pragma unroll
    for (int r = 0; r < 8; ++r)
      acc[r] += fast_tanh(c0[r]) * wr0[i] + fast_tanh(c1[r]) * wr16[i];
  };

  const int gw = blockIdx.x * NWAVES + wv;
  const int nw = gridDim.x * NWAVES;

  for (int t = gw; t < ntiles; t += nw) {
    const int base = t * 16;
    __builtin_prefetch(inp + (size_t)(base + nw * 16) * 5, 0, 0);  // next tile

    const float* ip = inp + (size_t)(base + nl) * 5;
    const float s0 = ip[0], s1 = ip[1], s2 = ip[2], s3 = ip[3], s4 = ip[4];
    const v16h fa = mkFeat(s0), fb = mkFeat(s1), fc = mkFeat(s2),
               fd = mkFeat(s3), fe = mkFeat(s4);

    float acc[8];
#pragma unroll
    for (int r = 0; r < 8; ++r) acc[r] = 0.0f;

    v16h M1, M2;
    M1 = message(fa, fb); M2 = message(fa, fd); update_ro(M1, M2, fa, true, 0, acc);
    M1 = message(fb, fa); M2 = message(fb, fc); update_ro(M1, M2, fb, true, 1, acc);
    M1 = message(fc, fb); M2 = message(fc, fd); update_ro(M1, M2, fc, true, 2, acc);
    M1 = message(fd, fc); M2 = message(fd, fa); update_ro(M1, M2, fd, true, 3, acc);
    update_ro(M1, M2, fe, false, 4, acc);  // Ue: messages are zero

    // Reduce over the 16 feature-columns (lanes within each half-group).
#pragma unroll
    for (int m = 1; m < 16; m <<= 1) {
#pragma unroll
      for (int r = 0; r < 8; ++r) acc[r] += __shfl_xor(acc[r], m, 32);
    }
    if (nl == 0) {  // lane 0 -> rows 0..7, lane 16 -> rows 8..15
      float4* o = reinterpret_cast<float4*>(out + base + (hi ? 8 : 0));
      o[0] = make_float4(acc[0] + brv, acc[1] + brv, acc[2] + brv, acc[3] + brv);
      o[1] = make_float4(acc[4] + brv, acc[5] + brv, acc[6] + brv, acc[7] + brv);
    }
  }
}

extern "C" void kernel_launch(void* const* d_in, const int* in_sizes, int n_in,
                              void* d_out, int out_size, void* d_ws, size_t ws_size,
                              hipStream_t stream) {
  (void)n_in; (void)d_ws; (void)ws_size; (void)out_size;
  const float* inp = (const float*)d_in[0];
  const float* Wf  = (const float*)d_in[1];
  const float* bf  = (const float*)d_in[2];
  const float* Wm  = (const float*)d_in[3];
  const float* bm  = (const float*)d_in[4];
  const float* Wu  = (const float*)d_in[5];
  const float* bu  = (const float*)d_in[6];
  const float* Wr  = (const float*)d_in[7];
  const float* br  = (const float*)d_in[8];
  float* out = (float*)d_out;

  const int B = in_sizes[0] / 5;
  const int ntiles = B / 16;  // B = 1048576 -> 65536 tiles
  mp17_kernel<<<dim3(2048), dim3(TPB), 0, stream>>>(
      inp, Wf, bf, Wm, bm, Wu, bu, Wr, br, out, ntiles);
}
